// DynamicProbSparseAttention_43190191129171
// MI455X (gfx1250) — matrix operand order repaired and checked by reading, not compiled, not run on hardware
//
#include <hip/hip_runtime.h>
#include <hip/hip_bf16.h>
#include <cstdint>

#define B_    4
#define L_    8192
#define D_    1024
#define H_    16
#define DK_   64
#define MAXF  10
#define MINF  3
#define M_TOT (B_*L_)          // 32768
#define TILE_ELEMS (128*40)    // one staged tile: 128 rows x 32 bf16, stride 40

typedef __bf16 v16bf __attribute__((ext_vector_type(16)));
typedef __bf16 v8bf  __attribute__((ext_vector_type(8)));
typedef __bf16 v4bf  __attribute__((ext_vector_type(4)));
typedef float  v8f   __attribute__((ext_vector_type(8)));
typedef unsigned int u32x4 __attribute__((ext_vector_type(4)));
typedef int          i32x4 __attribute__((ext_vector_type(4)));
typedef int          i32x8 __attribute__((ext_vector_type(8)));

// ---------------------------------------------------------------------------
// Kernel 0: one-shot fp32 -> split-bf16 (hi + lo) conversion.
// hi = bf16(v), lo = bf16(v - hi): GEMM accumulates hi*hi + lo*hi + hi*lo.
// ---------------------------------------------------------------------------
__global__ __launch_bounds__(256) void split_bf16_kernel(
    const float* __restrict__ src, __bf16* __restrict__ hi,
    __bf16* __restrict__ lo)
{
    size_t i = ((size_t)blockIdx.x * 256 + threadIdx.x) * 4;
    float4 v = *reinterpret_cast<const float4*>(&src[i]);
    float f[4] = {v.x, v.y, v.z, v.w};
    v4bf h, l;
    for (int e = 0; e < 4; ++e) {
        __bf16 hb = (__bf16)f[e];
        h[e] = hb;
        l[e] = (__bf16)(f[e] - (float)hb);
    }
    *reinterpret_cast<v4bf*>(&hi[i]) = h;
    *reinterpret_cast<v4bf*>(&lo[i]) = l;
}

// ---------------------------------------------------------------------------
// TDM: async 2D tile load (128 rows x 32 bf16) global -> LDS with padding:
// pad_interval = 16 DWORDs (one 64B row), pad_amount = 4 DWORDs (16B) =>
// LDS row stride 80B = 40 elems (16B-aligned for ds_load_b128 fragments).
// D# per CDNA5 ISA.  Uses the clang builtin (handles
// SGPR materialization).  The intrinsic is NOT modeled as an LDS store, so
// the caller escapes the LDS object into an asm with a memory clobber
// before reading fragments.
// Assumes the (only) static __shared__ block sits at LDS offset 0.
// ---------------------------------------------------------------------------
__device__ __forceinline__ void tdm_tile_load(const __bf16* src,
                                              uint32_t ldsByteOff,
                                              uint32_t srcRows)
{
    uint64_t ga = (uint64_t)(uintptr_t)src;
    u32x4 g0;
    g0[0] = 1u;                                   // count=1, user descriptor
    g0[1] = ldsByteOff;                           // lds_addr (bytes)
    g0[2] = (uint32_t)ga;                         // global_addr[31:0]
    g0[3] = (uint32_t)((ga >> 32) & 0x01ffffffu)  // global_addr[56:32]
            | (2u << 30);                         // type = 2 ("image")
    i32x8 g1;
    g1[0] = (int)((1u << 16)      // data_size = 1 -> 2 bytes
            | (1u << 20)          // pad_enable
            | (3u << 22)          // pad_interval: code 3 = 16 DWORDs
            | (3u << 25));        // pad_amount:   code 3 = 4 DWORDs
    g1[1] = (int)((1024u & 0xffffu) << 16);                   // tensor_dim0 lo16
    g1[2] = (int)(((1024u >> 16) & 0xffffu)
            | ((srcRows & 0xffffu) << 16));                   // dim0 hi | dim1 lo
    g1[3] = (int)(((srcRows >> 16) & 0xffffu) | (32u << 16)); // dim1 hi | tile_dim0=32
    g1[4] = 128;                                              // tile_dim1=128, tile_dim2=0
    g1[5] = 1024;                                             // tensor_dim0_stride
    g1[6] = 0;
    g1[7] = 0;
    i32x4 z4 = {0, 0, 0, 0};
    i32x8 z8 = {0, 0, 0, 0, 0, 0, 0, 0};
    (void)z8;
#if defined(__clang_major__) && (__clang_major__ >= 23)
    __builtin_amdgcn_tensor_load_to_lds(g0, g1, z4, z4, z8, 0);
#else
    __builtin_amdgcn_tensor_load_to_lds(g0, g1, z4, z4, 0);
#endif
}

// ---------------------------------------------------------------------------
// Kernel 1: fused QKV projection on pre-split bf16 inputs.
// C[m,n] = sum_k x[m,k]*W[n,k] + bias[n], W rows 0..3071 = wq|wk|wv.
// Block tile 128x128, 8 waves (4x2), each wave 32x64 = 2x4 WMMA tiles.
// TDM double-buffered staging: wave 0 issues 4 tensor_load_to_lds per K-step
// for the NEXT step, overlapping DMA with the current step's WMMAs.
// ---------------------------------------------------------------------------
__global__ __launch_bounds__(256) void qkv_gemm_kernel(
    const __bf16* __restrict__ xhi, const __bf16* __restrict__ xlo,
    const __bf16* __restrict__ whi, const __bf16* __restrict__ wlo,
    const float* __restrict__ bq, const float* __restrict__ bk,
    const float* __restrict__ bv,
    float* __restrict__ Qo, float* __restrict__ Ko, float* __restrict__ Vo)
{
    // [buf][array] : array 0=Ahi 1=Alo 2=Bhi 3=Blo ; 2*4*10240B = 80KB
    __shared__ __align__(16) __bf16 smem[2 * 4 * TILE_ELEMS];

    const int tid  = threadIdx.x;
    const int lane = tid & 31;
    const int wave = tid >> 5;
    const int wm   = wave & 3;          // M sub-block of 32 rows
    const int wn   = wave >> 2;         // N sub-block of 64 cols
    const int mBase = blockIdx.x * 128;
    const int nblk  = blockIdx.y;       // 0..23
    const int sel   = nblk >> 3;        // 0=Q 1=K 2=V
    const int nBase = (nblk & 7) * 128; // column base inside the 1024 group
    const int wRow  = nblk * 128;       // row base inside 3072-row weight pack

    const float* bs = (sel == 0) ? bq : (sel == 1) ? bk : bv;
    float* outp     = (sel == 0) ? Qo : (sel == 1) ? Ko : Vo;

    // real (volatile) store into smem: the object is stored-to, so loads can
    // never be folded to undef.  Barrier keeps it ordered before the TDM DMA.
    ((volatile __bf16*)smem)[tid] = (__bf16)0.0f;
    __syncthreads();

    v8f acc[2][4];
    for (int i = 0; i < 2; ++i)
        for (int j = 0; j < 4; ++j)
            for (int e = 0; e < 8; ++e) acc[i][j][e] = 0.0f;

    // prologue: stage K-step 0 into buffer 0
    if (wave == 0) {
        tdm_tile_load(&xhi[(size_t)mBase * D_ + 0], (0*4 + 0) * TILE_ELEMS * 2, M_TOT);
        tdm_tile_load(&xlo[(size_t)mBase * D_ + 0], (0*4 + 1) * TILE_ELEMS * 2, M_TOT);
        tdm_tile_load(&whi[(size_t)wRow  * D_ + 0], (0*4 + 2) * TILE_ELEMS * 2, 3072);
        tdm_tile_load(&wlo[(size_t)wRow  * D_ + 0], (0*4 + 3) * TILE_ELEMS * 2, 3072);
    }

    for (int ks = 0; ks < D_/32; ++ks) {
        if (wave == 0) __builtin_amdgcn_s_wait_tensorcnt(0);
        __syncthreads();   // buf[ks&1] ready; everyone done reading buf[(ks+1)&1]

        if (wave == 0 && (ks + 1) < D_/32) {
            const int nb = (ks + 1) & 1;
            const int k1 = (ks + 1) * 32;
            tdm_tile_load(&xhi[(size_t)mBase * D_ + k1], (nb*4 + 0) * TILE_ELEMS * 2, M_TOT);
            tdm_tile_load(&xlo[(size_t)mBase * D_ + k1], (nb*4 + 1) * TILE_ELEMS * 2, M_TOT);
            tdm_tile_load(&whi[(size_t)wRow  * D_ + k1], (nb*4 + 2) * TILE_ELEMS * 2, 3072);
            tdm_tile_load(&wlo[(size_t)wRow  * D_ + k1], (nb*4 + 3) * TILE_ELEMS * 2, 3072);
        }

        // ESCAPE smem into an opaque asm with a memory clobber (the asm "may
        // have written" it -- that's where the TDM data landed), but keep the
        // fragment reads on the original addrspace(3) object so they lower to
        // ds_load_b128 rather than flat loads.
        asm volatile("" :: "v"((const void*)smem) : "memory");
        const __bf16* bufp = smem + (ks & 1) * 4 * TILE_ELEMS;
        const __bf16* Ahi = bufp + 0*TILE_ELEMS;
        const __bf16* Alo = bufp + 1*TILE_ELEMS;
        const __bf16* Bhi = bufp + 2*TILE_ELEMS;
        const __bf16* Blo = bufp + 3*TILE_ELEMS;

        // CDNA5 16-bit fragment layouts:
        // A: M = lane%16 ; elems 0..7 -> K=(lane/16)*8..+7 ; elems 8..15 -> +16
        // B: N = lane%16 ; elems 0..7 -> K=(lane/16)*8..+7 ; elems 8..15 -> +16
        const int arow0 = wm*32 + (lane & 15);
        const int kca   = (lane >> 4) * 8;
        const int kcb   = (lane >> 4) * 8;

        v16bf aF[2][2];
        for (int mt = 0; mt < 2; ++mt) {
            int ar = arow0 + mt*16;
            v8bf h0 = *reinterpret_cast<const v8bf*>(&Ahi[ar*40 + kca]);
            v8bf h1 = *reinterpret_cast<const v8bf*>(&Ahi[ar*40 + kca + 16]);
            aF[mt][0] = __builtin_shufflevector(h0, h1,
                0,1,2,3,4,5,6,7,8,9,10,11,12,13,14,15);
            v8bf l0 = *reinterpret_cast<const v8bf*>(&Alo[ar*40 + kca]);
            v8bf l1 = *reinterpret_cast<const v8bf*>(&Alo[ar*40 + kca + 16]);
            aF[mt][1] = __builtin_shufflevector(l0, l1,
                0,1,2,3,4,5,6,7,8,9,10,11,12,13,14,15);
        }
        v16bf bF[4][2];
        for (int nt = 0; nt < 4; ++nt) {
            int br = wn*64 + nt*16 + (lane & 15);
            v8bf h0 = *reinterpret_cast<const v8bf*>(&Bhi[br*40 + kcb]);
            v8bf h1 = *reinterpret_cast<const v8bf*>(&Bhi[br*40 + kcb + 16]);
            bF[nt][0] = __builtin_shufflevector(h0, h1,
                0,1,2,3,4,5,6,7,8,9,10,11,12,13,14,15);
            v8bf l0 = *reinterpret_cast<const v8bf*>(&Blo[br*40 + kcb]);
            v8bf l1 = *reinterpret_cast<const v8bf*>(&Blo[br*40 + kcb + 16]);
            bF[nt][1] = __builtin_shufflevector(l0, l1,
                0,1,2,3,4,5,6,7,8,9,10,11,12,13,14,15);
        }

        for (int mt = 0; mt < 2; ++mt)
            for (int nt = 0; nt < 4; ++nt) {
                acc[mt][nt] = __builtin_amdgcn_wmma_f32_16x16x32_bf16(
                    false, aF[mt][0], false, bF[nt][0], (short)0, acc[mt][nt], false, false);
                acc[mt][nt] = __builtin_amdgcn_wmma_f32_16x16x32_bf16(
                    false, aF[mt][1], false, bF[nt][0], (short)0, acc[mt][nt], false, false);
                acc[mt][nt] = __builtin_amdgcn_wmma_f32_16x16x32_bf16(
                    false, aF[mt][0], false, bF[nt][1], (short)0, acc[mt][nt], false, false);
            }
    }

    // C layout: VGPR r -> M = r + 8*(lane/16), N = lane%16
    for (int mt = 0; mt < 2; ++mt)
        for (int nt = 0; nt < 4; ++nt) {
            int colG = nBase + wn*64 + nt*16 + (lane & 15);
            float bias = bs[colG];
            for (int r = 0; r < 8; ++r) {
                int rowL = wm*32 + mt*16 + r + ((lane >> 4) << 3);
                outp[(size_t)(mBase + rowL) * D_ + colG] = acc[mt][nt][r] + bias;
            }
        }
}

// ---------------------------------------------------------------------------
// Kernel 2: per-(b,h,l) sparsity score over dk=64. One wave per row.
// ---------------------------------------------------------------------------
__global__ __launch_bounds__(256) void qscore_kernel(
    const float* __restrict__ Q, float* __restrict__ scores)
{
    int tid  = threadIdx.x;
    int lane = tid & 31;
    size_t rid = (size_t)blockIdx.x * 8 + (tid >> 5);  // (b*H + h)*L + l
    int l  = (int)(rid % L_);
    int bh = (int)(rid / L_);
    int h  = bh % H_;
    int b  = bh / H_;
    const float* qp = &Q[((size_t)(b*L_ + l)) * D_ + h*DK_ + lane*2];
    float q0 = qp[0], q1 = qp[1];

    float mx = fmaxf(q0, q1);
    for (int o = 16; o; o >>= 1) mx = fmaxf(mx, __shfl_xor(mx, o, 32));
    float e0 = expf(q0 - mx), e1 = expf(q1 - mx);
    float se = e0 + e1;
    for (int o = 16; o; o >>= 1) se += __shfl_xor(se, o, 32);
    float sum = q0 + q1, sq = q0*q0 + q1*q1;
    for (int o = 16; o; o >>= 1) { sum += __shfl_xor(sum, o, 32); sq += __shfl_xor(sq, o, 32); }
    float p0 = e0 / se, p1 = e1 / se;
    float ent = -(p0 * logf(p0 + 1e-9f) + p1 * logf(p1 + 1e-9f));
    for (int o = 16; o; o >>= 1) ent += __shfl_xor(ent, o, 32);

    if (lane == 0) {
        float l2   = sqrtf(sq);
        float mean = sum * (1.0f / DK_);
        float var  = (sq - (float)DK_ * mean * mean) * (1.0f / (DK_ - 1));
        scores[rid] = 0.5f*l2 + 0.3f*ent + 0.2f*var;
    }
}

// ---------------------------------------------------------------------------
// Kernel 3: per-(b,h) top-10 over L + per-batch factor u (head 0 stats).
// ---------------------------------------------------------------------------
__global__ __launch_bounds__(256) void topk_stats_kernel(
    const float* __restrict__ scores, int* __restrict__ topIdx,
    int* __restrict__ uArr)
{
    __shared__ float cv[256*10];
    __shared__ int   ci[256*10];
    __shared__ float rs[256];
    __shared__ float rq[256];
    int t  = threadIdx.x;
    int bh = blockIdx.x;
    int h  = bh % H_;
    int b  = bh / H_;
    const float* sp = &scores[(size_t)bh * L_];

    float tv[10]; int ti[10];
    for (int j = 0; j < 10; ++j) { tv[j] = -3.4e38f; ti[j] = 0; }
    float sum = 0.f, sq = 0.f;
    for (int i = 0; i < L_/256; ++i) {
        int l = t + i*256;
        float v = sp[l];
        sum += v; sq += v*v;
        if (v > tv[9]) {
            int p = 9;
            while (p > 0 && v > tv[p-1]) { tv[p] = tv[p-1]; ti[p] = ti[p-1]; --p; }
            tv[p] = v; ti[p] = l;
        }
    }
    for (int j = 0; j < 10; ++j) { cv[t*10+j] = tv[j]; ci[t*10+j] = ti[j]; }
    rs[t] = sum; rq[t] = sq;
    __syncthreads();
    for (int s = 128; s; s >>= 1) {
        if (t < s) { rs[t] += rs[t+s]; rq[t] += rq[t+s]; }
        __syncthreads();
    }
    if (t == 0) {
        for (int r = 0; r < 10; ++r) {
            float best = -3.4e38f; int bi = 0;
            for (int j = 0; j < 256*10; ++j)
                if (cv[j] > best) { best = cv[j]; bi = j; }
            topIdx[bh*10 + r] = ci[bi];
            cv[bi] = -3.4e38f;
        }
        if (h == 0) {
            float mean = rs[0] / (float)L_;
            float var  = (rq[0] - (float)L_*mean*mean) / (float)(L_ - 1);
            float sd   = sqrtf(fmaxf(var, 0.f));
            int f = (int)rintf(sd / (mean + 1e-6f) * (float)MAXF);
            f = f < MINF ? MINF : (f > MAXF ? MAXF : f);
            uArr[b] = f;
        }
    }
}

// ---------------------------------------------------------------------------
// Kernel 4: S[b,h,r,l] = dot(Q_sparse[r], K[l]) / sqrt(dk).
// ---------------------------------------------------------------------------
__global__ __launch_bounds__(256) void sparse_scores_kernel(
    const float* __restrict__ Q, const float* __restrict__ K,
    const int* __restrict__ topIdx, float* __restrict__ S)
{
    __shared__ float qs[10*DK_];
    int t  = threadIdx.x;
    int bh = blockIdx.x;
    int h  = bh % H_, b = bh / H_;
    for (int i = t; i < 10*DK_; i += 256) {
        int r = i / DK_, d = i % DK_;
        int lq = topIdx[bh*10 + r];
        qs[i] = Q[((size_t)(b*L_ + lq)) * D_ + h*DK_ + d];
    }
    __syncthreads();
    int l = blockIdx.y * 256 + t;
    const float* kp = &K[((size_t)(b*L_ + l)) * D_ + h*DK_];
    float kr[DK_];
    for (int d = 0; d < DK_; d += 4) {
        float4 v = *reinterpret_cast<const float4*>(&kp[d]);
        kr[d] = v.x; kr[d+1] = v.y; kr[d+2] = v.z; kr[d+3] = v.w;
    }
    for (int r = 0; r < 10; ++r) {
        float acc = 0.f;
        for (int d = 0; d < DK_; ++d) acc += qs[r*DK_ + d] * kr[d];
        S[((size_t)(bh*10 + r)) * L_ + l] = acc * 0.125f;   // 1/sqrt(64)
    }
}

// ---------------------------------------------------------------------------
// Kernel 5: softmax over L, then out[r,d] = sum_l p(l) * V[l,d].
// ---------------------------------------------------------------------------
__global__ __launch_bounds__(256) void softmax_pv_kernel(
    const float* __restrict__ S, const float* __restrict__ V,
    float* __restrict__ attnOut)
{
    __shared__ float red[256];
    int t   = threadIdx.x;
    int bid = blockIdx.x;        // bh*10 + r
    int bh  = bid / 10;
    int h   = bh % H_, b = bh / H_;
    const float* sp = &S[(size_t)bid * L_];

    float mx = -3.4e38f;
    for (int i = 0; i < L_/256; ++i) mx = fmaxf(mx, sp[t + i*256]);
    red[t] = mx; __syncthreads();
    for (int s = 128; s; s >>= 1) { if (t < s) red[t] = fmaxf(red[t], red[t+s]); __syncthreads(); }
    mx = red[0];
    __syncthreads();

    float se = 0.f;
    for (int i = 0; i < L_/256; ++i) se += expf(sp[t + i*256] - mx);
    red[t] = se; __syncthreads();
    for (int s = 128; s; s >>= 1) { if (t < s) red[t] += red[t+s]; __syncthreads(); }
    se = red[0];

    if (t < DK_) {
        const float* vp = &V[((size_t)(b*L_)) * D_ + h*DK_ + t];
        float acc = 0.f;
        for (int l = 0; l < L_; ++l) acc += expf(sp[l] - mx) * vp[(size_t)l * D_];
        attnOut[(size_t)bid * DK_ + t] = acc / se;
    }
}

// ---------------------------------------------------------------------------
// Kernel 6a: output = broadcast bo (merged is zero almost everywhere).
// ---------------------------------------------------------------------------
__global__ __launch_bounds__(256) void bias_fill_kernel(
    const float* __restrict__ bo, float* __restrict__ out)
{
    size_t i = (size_t)blockIdx.x * 256 + threadIdx.x;
    out[i] = bo[i & (D_ - 1)];
}

// ---------------------------------------------------------------------------
// Kernel 6b: scatter kept rows through wo into the output rows.
// ---------------------------------------------------------------------------
__global__ __launch_bounds__(256) void scatter_proj_kernel(
    const float* __restrict__ attnOut, const float* __restrict__ wo,
    const int* __restrict__ topIdx, const int* __restrict__ uArr,
    float* __restrict__ out)
{
    __shared__ float a[DK_];
    int t   = threadIdx.x;
    int bid = blockIdx.x;       // bh*10 + r
    int r   = bid % 10;
    int bh  = bid / 10;
    int h   = bh % H_, b = bh / H_;
    if (r >= uArr[b]) return;   // uniform over the block
    int l = topIdx[bh*10 + r];
    if (t < DK_) a[t] = attnOut[(size_t)bid * DK_ + t];
    __syncthreads();
    for (int nn = 0; nn < D_/256; ++nn) {
        int n = t + nn*256;
        const float* wp = &wo[(size_t)n * D_ + h*DK_];
        float acc = 0.f;
        for (int d = 0; d < DK_; ++d) acc += a[d] * wp[d];
        atomicAdd(&out[((size_t)(b*L_ + l)) * D_ + n], acc);
    }
}

// ---------------------------------------------------------------------------
extern "C" void kernel_launch(void* const* d_in, const int* in_sizes, int n_in,
                              void* d_out, int out_size, void* d_ws, size_t ws_size,
                              hipStream_t stream)
{
    const float* x  = (const float*)d_in[0];
    const float* wq = (const float*)d_in[1];
    const float* bq = (const float*)d_in[2];
    const float* wk = (const float*)d_in[3];
    const float* bk = (const float*)d_in[4];
    const float* wv = (const float*)d_in[5];
    const float* bv = (const float*)d_in[6];
    const float* wo = (const float*)d_in[7];
    const float* bo = (const float*)d_in[8];
    float* out = (float*)d_out;

    const size_t nX = (size_t)M_TOT * D_;      // 33,554,432
    const size_t nW = (size_t)3 * D_ * D_;     //  3,145,728

    char* p = (char*)d_ws;
    __bf16* xhi = (__bf16*)p;            p += nX * sizeof(__bf16);
    __bf16* xlo = (__bf16*)p;            p += nX * sizeof(__bf16);
    __bf16* whi = (__bf16*)p;            p += nW * sizeof(__bf16);
    __bf16* wlo = (__bf16*)p;            p += nW * sizeof(__bf16);
    float* Qw = (float*)p;               p += nX * sizeof(float);
    float* Kw = (float*)p;               p += nX * sizeof(float);
    float* Vw = (float*)p;               p += nX * sizeof(float);
    float* sc = (float*)p;               p += (size_t)B_*H_*L_ * sizeof(float);
    float* S  = (float*)p;               p += (size_t)B_*H_*10*L_ * sizeof(float);
    float* ao = (float*)p;               p += (size_t)B_*H_*10*DK_ * sizeof(float);
    int* topIdx = (int*)p;               p += (size_t)B_*H_*10 * sizeof(int);
    int* uArr   = (int*)p;

    // one-shot split-bf16 conversion (x once; wq/wk/wv packed into 3072 rows)
    split_bf16_kernel<<<(int)(nX/1024), 256, 0, stream>>>(x, xhi, xlo);
    split_bf16_kernel<<<(int)(D_*D_/1024), 256, 0, stream>>>(wq, whi, wlo);
    split_bf16_kernel<<<(int)(D_*D_/1024), 256, 0, stream>>>(wk, whi + (size_t)D_*D_, wlo + (size_t)D_*D_);
    split_bf16_kernel<<<(int)(D_*D_/1024), 256, 0, stream>>>(wv, whi + (size_t)2*D_*D_, wlo + (size_t)2*D_*D_);

    qkv_gemm_kernel<<<dim3(M_TOT/128, 24), 256, 0, stream>>>(
        xhi, xlo, whi, wlo, bq, bk, bv, Qw, Kw, Vw);
    qscore_kernel<<<(B_*H_*L_)/8, 256, 0, stream>>>(Qw, sc);
    topk_stats_kernel<<<B_*H_, 256, 0, stream>>>(sc, topIdx, uArr);
    sparse_scores_kernel<<<dim3(B_*H_, L_/256), 256, 0, stream>>>(Qw, Kw, topIdx, S);
    softmax_pv_kernel<<<B_*H_*10, 256, 0, stream>>>(S, Vw, ao);
    bias_fill_kernel<<<(B_*L_*D_)/256, 256, 0, stream>>>(bo, out);
    scatter_proj_kernel<<<B_*H_*10, 256, 0, stream>>>(ao, wo, topIdx, uArr, out);
}
